// MultiHeadAttentionWithQKActivation_4002909520374
// MI455X (gfx1250) — compile-verified, matrix operand order checked by reading, not run on hardware
//
#include <hip/hip_runtime.h>
#include <hip/hip_bf16.h>
#include <math.h>

// ---------------- Types for WMMA ----------------
typedef __attribute__((ext_vector_type(16))) __bf16 v16bf;
typedef __attribute__((ext_vector_type(8)))  __bf16 v8bf;
typedef __attribute__((ext_vector_type(8)))  float  v8f;
typedef __attribute__((ext_vector_type(4)))  float  f32x4;

#define DIM     1024
#define NHEAD   16
#define HDIM    64
#define BATCH   2
#define SEQ     2048
#define MROWS   (BATCH * SEQ)          // 4096
#define SCALE   0.125f                 // 64^-0.5
#define CLIPTH  30.0f
#define NBINS   4096
#define HIST_LO 16.384f
#define HIST_IW 125.0f                 // 1 / 0.008
#define NTOT    134217728.0            // 2*16*2048*2048

// workspace layout (bytes)
#define OFF_Q   0
#define OFF_K   (8u  * 1024u * 1024u)
#define OFF_VT  (16u * 1024u * 1024u)
#define OFF_A   (24u * 1024u * 1024u)
#define OFF_S   (32u * 1024u * 1024u)
// stats u32 layout: [0..7]=4 doubles (s1..s4), [8]=enc max, [9..11]=c10,c30,c100,
// [12..15] pad, [16..16+NBINS) hist
#define STATS_U32 (16 + NBINS)

// ---------------- WMMA helper ----------------
__device__ __forceinline__ v8f wmma_bf16(v16bf a, v16bf b, v8f c) {
    return __builtin_amdgcn_wmma_f32_16x16x32_bf16(false, a, false, b, (short)0, c,
                                                   false, false);
}

// A/B fragment from row-major f32 tile (converted to bf16). ld in elements.
// Layout per ISA: lane<16 holds row (lane&15), K = {0..7, 16..23};
// lane>=16 holds same rows, K = {8..15, 24..31}.
__device__ __forceinline__ v16bf load_frag_f32(const float* tile, int ld) {
    const int lane = threadIdx.x & 31;
    const int r = lane & 15, hi = lane >> 4;
    const float* p = tile + r * ld + hi * 8;
    f32x4 a0 = *(const f32x4*)(p);
    f32x4 a1 = *(const f32x4*)(p + 4);
    f32x4 b0 = *(const f32x4*)(p + 16);
    f32x4 b1 = *(const f32x4*)(p + 20);
    v16bf f;
#pragma unroll
    for (int i = 0; i < 4; ++i) { f[i]      = (__bf16)a0[i]; f[i + 4]  = (__bf16)a1[i]; }
#pragma unroll
    for (int i = 0; i < 4; ++i) { f[i + 8]  = (__bf16)b0[i]; f[i + 12] = (__bf16)b1[i]; }
    return f;
}

// A/B fragment from row-major bf16 tile. ld in elements (multiple of 8).
__device__ __forceinline__ v16bf load_frag_bf16(const __bf16* tile, int ld) {
    const int lane = threadIdx.x & 31;
    const int r = lane & 15, hi = lane >> 4;
    const __bf16* p = tile + r * ld + hi * 8;
    v8bf lo = *(const v8bf*)(p);
    v8bf hi2 = *(const v8bf*)(p + 16);
    return __builtin_shufflevector(lo, hi2, 0,1,2,3,4,5,6,7,8,9,10,11,12,13,14,15);
}

// ordered-uint encoding for float atomic max
__device__ __forceinline__ unsigned fenc(float f) {
    unsigned u = __float_as_uint(f);
    return (u & 0x80000000u) ? ~u : (u | 0x80000000u);
}
__device__ __forceinline__ float fdec(unsigned u) {
    return (u & 0x80000000u) ? __uint_as_float(u & 0x7FFFFFFFu)
                             : __uint_as_float(~u);
}

// ---------------- kernels ----------------
__global__ void k_init(unsigned* stats) {
    int i = blockIdx.x * blockDim.x + threadIdx.x;
    if (i < STATS_U32) stats[i] = 0u;
}

// Fused QKV projection: C = x @ W^T, tanh on Q/K. One 16x64 output strip / wave.
__global__ void k_proj(const float* __restrict__ x,
                       const float* __restrict__ Wq,
                       const float* __restrict__ Wk,
                       const float* __restrict__ Wv,
                       __bf16* __restrict__ qws, __bf16* __restrict__ kws,
                       __bf16* __restrict__ vtws) {
    const int strip = blockIdx.x * 8 + (threadIdx.x >> 5);
    const int proj = strip / 4096;           // 0=Q 1=K 2=V
    const int s = strip - proj * 4096;
    const int m0 = (s >> 4) * 16;            // token-row tile
    const int o0 = (s & 15) * 64;            // out-col strip
    const float* W = (proj == 0) ? Wq : (proj == 1) ? Wk : Wv;

    v8f acc[4] = {};
    for (int kb = 0; kb < DIM; kb += 32) {
        v16bf a = load_frag_f32(x + m0 * DIM + kb, DIM);
        __builtin_prefetch(x + m0 * DIM + kb + 64, 0, 1);
#pragma unroll
        for (int t = 0; t < 4; ++t) {
            v16bf b = load_frag_f32(W + (o0 + t * 16) * DIM + kb, DIM);
            acc[t] = wmma_bf16(a, b, acc[t]);
        }
    }
    const int lane = threadIdx.x & 31;
    const int cn = lane & 15, hi = lane >> 4;
#pragma unroll
    for (int t = 0; t < 4; ++t) {
#pragma unroll
        for (int v = 0; v < 8; ++v) {
            const int rg = m0 + v + 8 * hi;          // global token row
            const int o  = o0 + t * 16 + cn;         // out channel
            const int b  = rg >> 11, n = rg & 2047;
            const int h  = o >> 6,  d = o & 63;
            const float val = acc[t][v];
            if (proj == 0)
                qws[((b * NHEAD + h) * SEQ + n) * HDIM + d] = (__bf16)tanhf(val);
            else if (proj == 1)
                kws[((b * NHEAD + h) * SEQ + n) * HDIM + d] = (__bf16)tanhf(val);
            else
                vtws[((b * NHEAD + h) * HDIM + d) * SEQ + n] = (__bf16)val;
        }
    }
}

// Telemetry pass: recompute scaled scores tile-by-tile, accumulate moments/max/
// threshold counts + LDS histogram.
__global__ void k_stats(const __bf16* __restrict__ qws,
                        const __bf16* __restrict__ kws,
                        unsigned* __restrict__ stats) {
    __shared__ unsigned hist[NBINS];
    for (int i = threadIdx.x; i < NBINS; i += 256) hist[i] = 0u;
    __syncthreads();

    const int bh = blockIdx.x >> 4;
    const int m0 = (blockIdx.x & 15) * 128 + (threadIdx.x >> 5) * 16;
    const __bf16* qb = qws + (bh * SEQ + m0) * HDIM;
    v16bf q0 = load_frag_bf16(qb, HDIM);
    v16bf q1 = load_frag_bf16(qb + 32, HDIM);

    double s1 = 0, s2 = 0, s3 = 0, s4 = 0;
    float mx = -3.4e38f;
    unsigned c10 = 0, c30 = 0, c100 = 0;

    for (int j = 0; j < SEQ; j += 16) {
        const __bf16* kb = kws + (bh * SEQ + j) * HDIM;
        __builtin_prefetch(kb + 16 * HDIM, 0, 1);
        v16bf k0 = load_frag_bf16(kb, HDIM);
        v16bf k1 = load_frag_bf16(kb + 32, HDIM);
        v8f c = {};
        c = wmma_bf16(q0, k0, c);
        c = wmma_bf16(q1, k1, c);
#pragma unroll
        for (int v = 0; v < 8; ++v) {
            const float sc = c[v] * SCALE;
            const double dx = (double)sc, d2 = dx * dx;
            s1 += dx; s2 += d2; s3 += d2 * dx; s4 += d2 * d2;
            mx = fmaxf(mx, sc);
            c10  += (sc > 10.0f);
            c30  += (sc > 30.0f);
            c100 += (sc > 100.0f);
            int bin = (int)((sc + HIST_LO) * HIST_IW);
            bin = min(max(bin, 0), NBINS - 1);
            atomicAdd(&hist[bin], 1u);
        }
    }
    // wave32 reduction
#pragma unroll
    for (int off = 16; off > 0; off >>= 1) {
        s1 += __shfl_down(s1, off);
        s2 += __shfl_down(s2, off);
        s3 += __shfl_down(s3, off);
        s4 += __shfl_down(s4, off);
        mx = fmaxf(mx, __shfl_down(mx, off));
        c10  += __shfl_down(c10, off);
        c30  += __shfl_down(c30, off);
        c100 += __shfl_down(c100, off);
    }
    if ((threadIdx.x & 31) == 0) {
        double* S = (double*)stats;
        atomicAdd(&S[0], s1); atomicAdd(&S[1], s2);
        atomicAdd(&S[2], s3); atomicAdd(&S[3], s4);
        atomicMax(&stats[8], fenc(mx));
        atomicAdd(&stats[9], c10);
        atomicAdd(&stats[10], c30);
        atomicAdd(&stats[11], c100);
    }
    __syncthreads();
    for (int i = threadIdx.x; i < NBINS; i += 256)
        if (hist[i]) atomicAdd(&stats[16 + i], hist[i]);
}

__global__ void k_finalize(const unsigned* __restrict__ stats,
                           float* __restrict__ outs) {
    if (threadIdx.x != 0 || blockIdx.x != 0) return;
    const double* S = (const double*)stats;
    const double s1 = S[0], s2 = S[1], s3 = S[2], s4 = S[3];
    const double mean = s1 / NTOT;
    double var = (s2 - NTOT * mean * mean) / (NTOT - 1.0);
    if (var < 0.0) var = 0.0;
    const double sd = sqrt(var);
    const double m4 = s4 / NTOT - 4.0 * mean * s3 / NTOT +
                      6.0 * mean * mean * s2 / NTOT - 3.0 * mean * mean * mean * mean;
    const double kurt = (sd > 1e-8) ? (m4 / (sd * sd * sd * sd) - 3.0) : 0.0;
    const float mx = fdec(stats[8]);

    const double qv[4] = {0.5, 0.9, 0.99, 0.999};
    float qs[4];
#pragma unroll
    for (int t = 0; t < 4; ++t) {
        const double rank = qv[t] * (NTOT - 1.0);
        double cum = 0.0;
        float val = HIST_LO;
        for (int i = 0; i < NBINS; ++i) {
            const double c = (double)stats[16 + i];
            if (cum + c > rank) {
                const double frac = (rank - cum) / (c > 1.0 ? c : 1.0);
                val = (float)(-(double)HIST_LO + ((double)i + frac) * 0.008);
                break;
            }
            cum += c;
        }
        qs[t] = val;
    }
    outs[0] = mx;
    outs[1] = (float)mean;
    outs[2] = (float)sd;
    outs[3] = qs[0]; outs[4] = qs[1]; outs[5] = qs[2]; outs[6] = qs[3];
    outs[7] = (float)kurt;
    outs[8]  = (float)((double)stats[9]  / NTOT);
    outs[9]  = (float)((double)stats[10] / NTOT);
    outs[10] = (float)((double)stats[11] / NTOT);
    outs[11] = (float)((double)stats[10] / NTOT);        // clip_fraction (>30)
    outs[12] = (mx > CLIPTH) ? CLIPTH : mx;              // max_post_clip
}

// Flash attention: online softmax, O = softmax(clip(QK^T*s)) @ V.
__global__ void k_attn(const __bf16* __restrict__ qws,
                       const __bf16* __restrict__ kws,
                       const __bf16* __restrict__ vtws,
                       __bf16* __restrict__ aws,
                       const unsigned* __restrict__ stats) {
    __shared__ __align__(16) __bf16 plds[8][16][32];
    const int bh = blockIdx.x >> 4;
    const int wv = threadIdx.x >> 5;
    const int m0 = (blockIdx.x & 15) * 128 + wv * 16;
    const int lane = threadIdx.x & 31;
    const int cn = lane & 15, hi = lane >> 4;
    const bool doclip = fdec(stats[8]) > CLIPTH;

    const __bf16* qb = qws + (bh * SEQ + m0) * HDIM;
    v16bf q0 = load_frag_bf16(qb, HDIM);
    v16bf q1 = load_frag_bf16(qb + 32, HDIM);

    v8f o[4] = {};
    float mrow[8], lrow[8];
#pragma unroll
    for (int v = 0; v < 8; ++v) { mrow[v] = -1e30f; lrow[v] = 0.0f; }

    for (int j = 0; j < SEQ; j += 32) {
        const __bf16* kb = kws + (bh * SEQ + j) * HDIM;
        __builtin_prefetch(kb + 32 * HDIM, 0, 1);
        v16bf ka0 = load_frag_bf16(kb, HDIM);
        v16bf ka1 = load_frag_bf16(kb + 32, HDIM);
        v16bf kb0 = load_frag_bf16(kb + 16 * HDIM, HDIM);
        v16bf kb1 = load_frag_bf16(kb + 16 * HDIM + 32, HDIM);
        v8f s0 = {}, s1 = {};
        s0 = wmma_bf16(q0, ka0, s0); s0 = wmma_bf16(q1, ka1, s0);
        s1 = wmma_bf16(q0, kb0, s1); s1 = wmma_bf16(q1, kb1, s1);

        float p0[8], p1[8], rm[8];
#pragma unroll
        for (int v = 0; v < 8; ++v) {
            float a = s0[v] * SCALE, b = s1[v] * SCALE;
            if (doclip) { a = fminf(a, CLIPTH); b = fminf(b, CLIPTH); }
            p0[v] = a; p1[v] = b;
            rm[v] = fmaxf(a, b);
        }
        // row max across the 16 column lanes of each half
#pragma unroll
        for (int msk = 1; msk < 16; msk <<= 1)
#pragma unroll
            for (int v = 0; v < 8; ++v)
                rm[v] = fmaxf(rm[v], __shfl_xor(rm[v], msk));

        float sf[8], rs[8];
#pragma unroll
        for (int v = 0; v < 8; ++v) {
            const float mn = fmaxf(mrow[v], rm[v]);
            sf[v] = __expf(mrow[v] - mn);
            mrow[v] = mn;
            p0[v] = __expf(p0[v] - mn);
            p1[v] = __expf(p1[v] - mn);
            rs[v] = p0[v] + p1[v];
        }
#pragma unroll
        for (int msk = 1; msk < 16; msk <<= 1)
#pragma unroll
            for (int v = 0; v < 8; ++v)
                rs[v] += __shfl_xor(rs[v], msk);
#pragma unroll
        for (int v = 0; v < 8; ++v) lrow[v] = lrow[v] * sf[v] + rs[v];
#pragma unroll
        for (int t = 0; t < 4; ++t)
#pragma unroll
            for (int v = 0; v < 8; ++v) o[t][v] *= sf[v];

        // restage P (16x32) into per-wave LDS, then re-read in A-frag layout
#pragma unroll
        for (int v = 0; v < 8; ++v) {
            const int r = v + 8 * hi;
            plds[wv][r][cn]      = (__bf16)p0[v];
            plds[wv][r][16 + cn] = (__bf16)p1[v];
        }
        asm volatile("s_wait_dscnt 0" ::: "memory");
        v16bf pa = load_frag_bf16(&plds[wv][0][0], 32);
#pragma unroll
        for (int t = 0; t < 4; ++t) {
            v16bf vb = load_frag_bf16(vtws + (bh * HDIM + t * 16) * SEQ + j, SEQ);
            o[t] = wmma_bf16(pa, vb, o[t]);
        }
    }

    float inv[8];
#pragma unroll
    for (int v = 0; v < 8; ++v) inv[v] = 1.0f / lrow[v];
    const int b = bh >> 4, h = bh & 15;
#pragma unroll
    for (int t = 0; t < 4; ++t)
#pragma unroll
        for (int v = 0; v < 8; ++v) {
            const int n = m0 + v + 8 * hi;
            const int d = t * 16 + cn;
            aws[(b * SEQ + n) * DIM + h * HDIM + d] = (__bf16)(o[t][v] * inv[v]);
        }
}

// Output projection: out = A @ Wo^T + bo (f32 out)
__global__ void k_oproj(const __bf16* __restrict__ aws,
                        const float* __restrict__ Wo,
                        const float* __restrict__ bo,
                        float* __restrict__ out) {
    const int strip = blockIdx.x * 8 + (threadIdx.x >> 5);
    const int m0 = (strip >> 4) * 16;
    const int o0 = (strip & 15) * 64;
    v8f acc[4] = {};
    for (int kb = 0; kb < DIM; kb += 32) {
        v16bf a = load_frag_bf16(aws + m0 * DIM + kb, DIM);
#pragma unroll
        for (int t = 0; t < 4; ++t) {
            v16bf b = load_frag_f32(Wo + (o0 + t * 16) * DIM + kb, DIM);
            acc[t] = wmma_bf16(a, b, acc[t]);
        }
    }
    const int lane = threadIdx.x & 31;
    const int cn = lane & 15, hi = lane >> 4;
#pragma unroll
    for (int t = 0; t < 4; ++t)
#pragma unroll
        for (int v = 0; v < 8; ++v) {
            const int rg = m0 + v + 8 * hi;
            const int oc = o0 + t * 16 + cn;
            out[rg * DIM + oc] = acc[t][v] + bo[oc];
        }
}

// ---------------- launch ----------------
extern "C" void kernel_launch(void* const* d_in, const int* in_sizes, int n_in,
                              void* d_out, int out_size, void* d_ws, size_t ws_size,
                              hipStream_t stream) {
    const float* x  = (const float*)d_in[0];
    const float* Wq = (const float*)d_in[1];
    const float* Wk = (const float*)d_in[2];
    const float* Wv = (const float*)d_in[3];
    const float* Wo = (const float*)d_in[4];
    const float* bo = (const float*)d_in[5];
    float* out = (float*)d_out;

    char* w = (char*)d_ws;
    __bf16* qws  = (__bf16*)(w + OFF_Q);
    __bf16* kws  = (__bf16*)(w + OFF_K);
    __bf16* vtws = (__bf16*)(w + OFF_VT);
    __bf16* aws  = (__bf16*)(w + OFF_A);
    unsigned* stats = (unsigned*)(w + OFF_S);

    k_init<<<(STATS_U32 + 255) / 256, 256, 0, stream>>>(stats);
    k_proj<<<1536, 256, 0, stream>>>(x, Wq, Wk, Wv, qws, kws, vtws);
    k_stats<<<512, 256, 0, stream>>>(qws, kws, stats);
    k_finalize<<<1, 32, 0, stream>>>(stats, out + (size_t)MROWS * DIM);
    k_attn<<<512, 256, 0, stream>>>(qws, kws, vtws, aws, stats);
    k_oproj<<<512, 256, 0, stream>>>(aws, Wo, bo, out);
}
